// GraphGenerativeModel_3324304687517
// MI455X (gfx1250) — matrix operand order repaired and checked by reading, not compiled, not run on hardware
//
#include <hip/hip_runtime.h>
#include <hip/hip_bf16.h>
#include <math.h>

// ---------------------------------------------------------------------------
// GCN pipeline for gfx1250 (MI455X): WMMA bf16 GEMMs + atomic edge scatter.
// ---------------------------------------------------------------------------

typedef __attribute__((ext_vector_type(16))) __bf16 v16bf;
typedef __attribute__((ext_vector_type(8)))  float  v8f;

// ---------------------------------------------------------------------------
// Utility: zero a float region (graph-capture safe, no hipMemset)
// ---------------------------------------------------------------------------
__global__ __launch_bounds__(256) void zero_f32(float* __restrict__ p, long long n) {
  long long i = blockIdx.x * (long long)blockDim.x + threadIdx.x;
  long long stride = (long long)gridDim.x * blockDim.x;
  for (; i < n; i += stride) p[i] = 0.0f;
}

// ---------------------------------------------------------------------------
// deg[c] += ew[e]  (scatter over col)
// ---------------------------------------------------------------------------
__global__ __launch_bounds__(256) void deg_scatter(const int* __restrict__ colI,
                                                   const float* __restrict__ ew,
                                                   float* __restrict__ deg, int E) {
  int e = blockIdx.x * blockDim.x + threadIdx.x;
  if (e >= E) return;
  __hip_atomic_fetch_add(&deg[colI[e]], ew[e], __ATOMIC_RELAXED, __HIP_MEMORY_SCOPE_AGENT);
}

// dis[n] = rsqrt(deg[n] + 1)   (in place; deg >= 0 so deg+1 >= 1, no inf case)
__global__ __launch_bounds__(256) void make_dis(float* __restrict__ deg_dis, int N) {
  int n = blockIdx.x * blockDim.x + threadIdx.x;
  if (n >= N) return;
  deg_dis[n] = rsqrtf(deg_dis[n] + 1.0f);
}

// ---------------------------------------------------------------------------
// WMMA GEMM: Out[M x NCOLS] = A[M x KDIM] (f32, row major) * W[KDIM x NCOLS]
// A,W converted to bf16; f32 accumulate via v_wmma_f32_16x16x32_bf16.
//
// W is staged once per block into LDS *pre-swizzled into B-fragment layout*:
//   Wlds[((kt*NT + t)*32 + lane)*16 + i]  holds W[kt*32 + (lane>>4)*16 + i][t*16 + (lane&15)]
// so the inner-loop B fragment is a single 32-byte-aligned v16bf load per lane
// (2x ds_load_b128) instead of 16 scalar ds_load_u16.
//
// One wave computes one 16-row M tile across all NCOLS (NT n-tiles), reusing
// the A fragment across the n-tiles.
// Requires: M % 16 == 0, KDIM % 32 == 0, NCOLS % 16 == 0.
// ---------------------------------------------------------------------------
template <int KDIM, int NCOLS>
__global__ __launch_bounds__(256) void wmma_gemm(const float* __restrict__ A,
                                                 const float* __restrict__ W,
                                                 float* __restrict__ Out, int M) {
  constexpr int NT = NCOLS / 16;
  constexpr int KT = KDIM / 32;
  __shared__ __attribute__((aligned(32))) __bf16 Wlds[KDIM * NCOLS];

  // Stage W -> LDS, swizzled into fragment-major layout (coalesced global reads).
  for (int idx = threadIdx.x; idx < KDIM * NCOLS; idx += blockDim.x) {
    const int k = idx >> (NCOLS == 128 ? 7 : 6);
    const int n = idx & (NCOLS - 1);
    const int kt = k >> 5;
    const int kh = (k >> 4) & 1;
    const int i  = k & 15;
    const int slot = ((kt * NT + (n >> 4)) * 32 + kh * 16 + (n & 15)) * 16 + i;
    Wlds[slot] = (__bf16)W[idx];
  }
  __syncthreads();

  const int lane     = threadIdx.x & 31;
  const int laneHalf = lane >> 4;   // 0: lanes 0-15, 1: lanes 16-31
  const int lmod     = lane & 15;
  const int mtile = blockIdx.x * (blockDim.x >> 5) + (threadIdx.x >> 5);
  const int m0 = mtile * 16;
  if (m0 >= M) return;              // wave-uniform: EXEC stays all-1s for WMMA

  v8f acc[NT];
#pragma unroll
  for (int t = 0; t < NT; ++t) acc[t] = (v8f){0.f,0.f,0.f,0.f,0.f,0.f,0.f,0.f};

  const float*  arow  = A + (size_t)(m0 + lmod) * KDIM;
  const v16bf*  wfrag = (const v16bf*)Wlds;   // indexed by (kt*NT + t)*32 + lane

  for (int kt = 0; kt < KT; ++kt) {
    const int kb = kt * 32;
    // A fragment (contiguous k order, same as the LDS B fragments):
    //   lane<16 : elem i -> k = kb + i
    //   lane>=16: elem i -> k = kb + 16 + i
    v16bf afrag;
#pragma unroll
    for (int i = 0; i < 16; ++i) {
      int k = kb + laneHalf * 16 + i;
      afrag[i] = (__bf16)arow[k];
    }
    const v16bf* wk = wfrag + (size_t)kt * NT * 32 + lane;
#pragma unroll
    for (int t = 0; t < NT; ++t) {
      v16bf bfrag = wk[t * 32];
      acc[t] = __builtin_amdgcn_wmma_f32_16x16x32_bf16(
          false, afrag, false, bfrag, (short)0, acc[t], false, false);
    }
  }

  // D layout: VGPR r -> row m0 + r + laneHalf*8, col = n-tile base + lmod
#pragma unroll
  for (int t = 0; t < NT; ++t) {
    const int n = t * 16 + lmod;
#pragma unroll
    for (int r = 0; r < 8; ++r)
      Out[(size_t)(m0 + r + laneHalf * 8) * NCOLS + n] = acc[t][r];
  }
}

// ---------------------------------------------------------------------------
// Edge scatter: agg[col[e]][f] += xw[row[e]][f] * (dis[row]*ew[e]*dis[col])
// One thread per (edge, feature); f32 global atomics, coalesced over f.
// ---------------------------------------------------------------------------
template <int F>
__global__ __launch_bounds__(256) void edge_scatter(const int* __restrict__ rowI,
                                                    const int* __restrict__ colI,
                                                    const float* __restrict__ ew,
                                                    const float* __restrict__ dis,
                                                    const float* __restrict__ xw,
                                                    float* __restrict__ agg, int E) {
  long long t = blockIdx.x * (long long)blockDim.x + threadIdx.x;
  if (t >= (long long)E * F) return;
  const int e = (int)(t / F);
  const int f = (int)(t - (long long)e * F);
  const int r = rowI[e];
  const int c = colI[e];
  const float nrm = dis[r] * ew[e] * dis[c];
  const float v = xw[(size_t)r * F + f] * nrm;
  __hip_atomic_fetch_add(&agg[(size_t)c * F + f], v,
                         __ATOMIC_RELAXED, __HIP_MEMORY_SCOPE_AGENT);
}

// ---------------------------------------------------------------------------
// h = relu(agg + xw * dis^2 + b)   (self-loop 1/deg term + bias); in place.
// ---------------------------------------------------------------------------
template <int F>
__global__ __launch_bounds__(256) void combine_relu(float* __restrict__ agg_h,
                                                    const float* __restrict__ xw,
                                                    const float* __restrict__ dis,
                                                    const float* __restrict__ b, int N) {
  long long t = blockIdx.x * (long long)blockDim.x + threadIdx.x;
  if (t >= (long long)N * F) return;
  const int n = (int)(t / F);
  const int f = (int)(t - (long long)n * F);
  const float d = dis[n];
  float v = agg_h[t] + xw[t] * d * d + b[f];
  agg_h[t] = v > 0.f ? v : 0.f;
}

// ---------------------------------------------------------------------------
// Layer-2 combine + global mean-pool accumulation (sum + count).
// ---------------------------------------------------------------------------
__global__ __launch_bounds__(256) void combine2_pool(const float* __restrict__ agg2,
                                                     const float* __restrict__ xw2,
                                                     const float* __restrict__ dis,
                                                     const float* __restrict__ b2,
                                                     const int* __restrict__ batch,
                                                     float* __restrict__ psum,
                                                     float* __restrict__ pcnt, int N) {
  long long t = blockIdx.x * (long long)blockDim.x + threadIdx.x;
  if (t >= (long long)N * 64) return;
  const int n = (int)(t >> 6);
  const int f = (int)(t & 63);
  const float d = dis[n];
  float v = agg2[t] + xw2[t] * d * d + b2[f];
  v = v > 0.f ? v : 0.f;
  const int g = batch[n];
  __hip_atomic_fetch_add(&psum[g * 64 + f], v,
                         __ATOMIC_RELAXED, __HIP_MEMORY_SCOPE_AGENT);
  if (f == 0)
    __hip_atomic_fetch_add(&pcnt[g], 1.0f,
                           __ATOMIC_RELAXED, __HIP_MEMORY_SCOPE_AGENT);
}

// g = psum / max(pcnt, 1)
__global__ __launch_bounds__(256) void pool_mean(const float* __restrict__ psum,
                                                 const float* __restrict__ pcnt,
                                                 float* __restrict__ g) {
  int i = blockIdx.x * blockDim.x + threadIdx.x;
  if (i >= 64 * 64) return;
  g[i] = psum[i] / fmaxf(pcnt[i >> 6], 1.0f);
}

// ---------------------------------------------------------------------------
// Final FC: Out[64 x NOUT] = G[64 x 64] @ Wfc[64 x NOUT] + bfc.
// NOUT = 2904 is not a multiple of 16. Out-of-range columns are handled by
// CLAMPING the load index (always in-bounds -> unconditional loads, no EXEC
// juggling) and selecting the loaded VALUE to zero (v_cndmask). Stores are
// masked per-lane after the WMMAs.
// ---------------------------------------------------------------------------
__global__ __launch_bounds__(256) void wmma_fc(const float* __restrict__ G,
                                               const float* __restrict__ Wfc,
                                               const float* __restrict__ bfc,
                                               float* __restrict__ Out, int NOUT) {
  const int lane     = threadIdx.x & 31;
  const int laneHalf = lane >> 4;
  const int lmod     = lane & 15;
  const int ntiles = (NOUT + 15) >> 4;
  const int tile = blockIdx.x * (blockDim.x >> 5) + (threadIdx.x >> 5);
  if (tile >= 4 * ntiles) return;          // wave-uniform
  const int m0 = (tile & 3) * 16;          // M = 64 -> 4 tiles
  const int nb = (tile >> 2) * 16;
  const int n  = nb + lmod;
  const bool nvalid = (n < NOUT);
  const int  ncl = nvalid ? n : (NOUT - 1);   // clamped, always in-bounds

  v8f acc = (v8f){0.f,0.f,0.f,0.f,0.f,0.f,0.f,0.f};
  const float* grow = G + (size_t)(m0 + lmod) * 64;
  const float* wcol = Wfc + ncl;

#pragma unroll
  for (int kb = 0; kb < 64; kb += 32) {
    v16bf afrag, bfrag;
#pragma unroll
    for (int i = 0; i < 16; ++i) {
      int k = kb + laneHalf * 16 + i;
      afrag[i] = (__bf16)grow[k];
    }
#pragma unroll
    for (int i = 0; i < 16; ++i) {
      int k = kb + laneHalf * 16 + i;
      float wv = wcol[(size_t)k * NOUT];       // unconditional, affine in k
      bfrag[i] = (__bf16)(nvalid ? wv : 0.f);  // select on value, not address
    }
    acc = __builtin_amdgcn_wmma_f32_16x16x32_bf16(
        false, afrag, false, bfrag, (short)0, acc, false, false);
  }

  if (nvalid) {
    const float bias = bfc[n];
#pragma unroll
    for (int r = 0; r < 8; ++r)
      Out[(size_t)(m0 + r + laneHalf * 8) * NOUT + n] = acc[r] + bias;
  }
}

// ---------------------------------------------------------------------------
// Launch: setup_inputs order:
// 0:x[N,128] 1:edge_index[2,E] 2:edge_weight[E] 3:batch[N]
// 4:W1[128,128] 5:b1[128] 6:W2[128,64] 7:b2[64] 8:Wfc[64,2904] 9:bfc[2904]
// out: [64, 2904] f32
// ---------------------------------------------------------------------------
extern "C" void kernel_launch(void* const* d_in, const int* in_sizes, int n_in,
                              void* d_out, int out_size, void* d_ws, size_t ws_size,
                              hipStream_t stream) {
  const float* x     = (const float*)d_in[0];
  const int*   ei    = (const int*)d_in[1];
  const float* ew    = (const float*)d_in[2];
  const int*   batch = (const int*)d_in[3];
  const float* W1    = (const float*)d_in[4];
  const float* b1    = (const float*)d_in[5];
  const float* W2    = (const float*)d_in[6];
  const float* b2    = (const float*)d_in[7];
  const float* Wfc   = (const float*)d_in[8];
  const float* bfc   = (const float*)d_in[9];
  float* out = (float*)d_out;

  const int N    = in_sizes[0] / 128;   // 40000
  const int E    = in_sizes[2];         // 640000
  const int FLAT = in_sizes[9];         // 2904
  const int* rowI = ei;
  const int* colI = ei + E;

  // Workspace layout (floats): deg/dis | agg1(->h) | xw1 (xw2/agg2 alias) | psum|pcnt|g
  float* deg_dis = (float*)d_ws;                        // N
  float* agg1    = deg_dis + N;                         // N*128  (becomes h)
  float* xw1     = agg1 + (size_t)N * 128;              // N*128
  float* xw2     = xw1;                                 // N*64  (after xw1 is dead)
  float* agg2    = xw1 + (size_t)N * 64;                // N*64  (after xw1 is dead)
  float* psum    = xw1 + (size_t)N * 128;               // 64*64
  float* pcnt    = psum + 64 * 64;                      // 64
  float* gbuf    = pcnt + 64;                           // 64*64

  const dim3 blk(256);
  auto nb = [](long long n) { return (unsigned)((n + 255) / 256); };

  // Zero deg + agg1 (contiguous), and pool accumulators.
  zero_f32<<<nb((long long)N * 129), blk, 0, stream>>>(deg_dis, (long long)N * 129);
  zero_f32<<<nb(64 * 64 + 64), blk, 0, stream>>>(psum, 64 * 64 + 64);

  // Degree / normalization.
  deg_scatter<<<nb(E), blk, 0, stream>>>(colI, ew, deg_dis, E);
  make_dis<<<nb(N), blk, 0, stream>>>(deg_dis, N);

  // Layer 1: xw1 = x @ W1  (WMMA bf16)
  {
    const int mtiles = (N + 15) / 16;
    wmma_gemm<128, 128><<<(mtiles + 7) / 8, blk, 0, stream>>>(x, W1, xw1, N);
  }
  edge_scatter<128><<<nb((long long)E * 128), blk, 0, stream>>>(rowI, colI, ew, deg_dis, xw1, agg1, E);
  combine_relu<128><<<nb((long long)N * 128), blk, 0, stream>>>(agg1, xw1, deg_dis, b1, N);

  // Layer 2: xw2 = h @ W2  (h lives in agg1; xw2 aliases dead xw1 region)
  {
    const int mtiles = (N + 15) / 16;
    wmma_gemm<128, 64><<<(mtiles + 7) / 8, blk, 0, stream>>>(agg1, W2, xw2, N);
  }
  zero_f32<<<nb((long long)N * 64), blk, 0, stream>>>(agg2, (long long)N * 64);
  edge_scatter<64><<<nb((long long)E * 64), blk, 0, stream>>>(rowI, colI, ew, deg_dis, xw2, agg2, E);
  combine2_pool<<<nb((long long)N * 64), blk, 0, stream>>>(agg2, xw2, deg_dis, b2, batch, psum, pcnt, N);

  // Mean pool + FC head.
  pool_mean<<<nb(64 * 64), blk, 0, stream>>>(psum, pcnt, gbuf);
  {
    const int ntiles = (FLAT + 15) / 16;
    const int tiles = 4 * ntiles;
    wmma_fc<<<(tiles + 7) / 8, blk, 0, stream>>>(gbuf, Wfc, bfc, out, FLAT);
  }
}
